// GraphTransformer_21182778704875
// MI455X (gfx1250) — matrix operand order repaired and checked by reading, not compiled, not run on hardware
//
#include <hip/hip_runtime.h>

typedef __attribute__((ext_vector_type(16))) _Float16 v16h;
typedef __attribute__((ext_vector_type(8)))  float    v8f;

__device__ __forceinline__ float lrelu(float x){ return x > 0.f ? x : 0.2f * x; }

// monotonic float <-> uint mapping for atomicMax on floats
__device__ __forceinline__ unsigned mapf(float x){
  unsigned u = __float_as_uint(x);
  return (u & 0x80000000u) ? ~u : (u | 0x80000000u);
}
__device__ __forceinline__ float unmapf(unsigned m){
  return __uint_as_float((m & 0x80000000u) ? (m & 0x7FFFFFFFu) : ~m);
}

__global__ void k_fill_f32(float* p, float v, int n){
  int i = blockIdx.x * blockDim.x + threadIdx.x;
  if (i < n) p[i] = v;
}
__global__ void k_fill_u32(unsigned* p, unsigned v, int n){
  int i = blockIdx.x * blockDim.x + threadIdx.x;
  if (i < n) p[i] = v;
}

// ---- self-loop mean edge_attr: segment-sum + count over original edges ----
// one thread per (edge, 4-feature group): 1x b128 load + 4 atomics
__global__ void k_mean_accum(const float* __restrict__ ea, const int* __restrict__ dst,
                             float* sums, float* cnt, int E){
  int t = blockIdx.x * blockDim.x + threadIdx.x;
  if (t >= E * 4) return;
  int e = t >> 2, f4 = t & 3;
  int d = dst[e];
  const float4 v = *(const float4*)(ea + (size_t)e * 16 + f4 * 4);
  float* sp = sums + (size_t)d * 16 + f4 * 4;
  atomicAdd(sp + 0, v.x);
  atomicAdd(sp + 1, v.y);
  atomicAdd(sp + 2, v.z);
  atomicAdd(sp + 3, v.w);
  if (f4 == 0) atomicAdd(&cnt[d], 1.0f);
}
__global__ void k_mean_div(float* sums, const float* __restrict__ cnt, int N){
  int t = blockIdx.x * blockDim.x + threadIdx.x;
  if (t >= N * 16) return;
  sums[t] = sums[t] / fmaxf(cnt[t >> 4], 1.0f);
}

// ---- WMMA GEMM: C[M,Nc] = act(A[M,K] @ B[K,Nc] + bias). One wave per 16x16 tile.
// f32 -> f16 on the fly, f32 accumulate via v_wmma_f32_16x16x32_f16.
template<bool RELU, bool BIAS>
__global__ __launch_bounds__(32) void k_wmma_gemm(const float* __restrict__ A,
                                                  const float* __restrict__ B,
                                                  const float* __restrict__ bias,
                                                  float* __restrict__ C,
                                                  int K, int Nc){
  const int lane = threadIdx.x;       // full wave, EXEC all ones
  const int l16  = lane & 15;
  const int half = lane >> 4;
  const int rt   = blockIdx.x;
  const int col  = blockIdx.y * 16 + l16;
  v8f acc = {};
  const float* arow = A + (size_t)(rt * 16 + l16) * K;
  for (int kk = 0; kk < K; kk += 32) {
    v16h a, b;
    // A 16x32 f16, contiguous k (matches B): lane<16 row M=l16 holds K=0..15; lane>=16 K=16..31
#pragma unroll
    for (int v = 0; v < 8; ++v) {
      int kb = kk + (half << 4) + (v << 1);
      a[2*v]   = (_Float16)arow[kb];
      a[2*v+1] = (_Float16)arow[kb + 1];
    }
    // B 32x16 f16, contiguous k (matches A): lanes 0-15 col N, K=0..15; lanes 16-31 col N, K=16..31
    const float* bcol = B + (size_t)(kk + (half << 4)) * Nc + col;
#pragma unroll
    for (int v = 0; v < 8; ++v) {
      b[2*v]   = (_Float16)bcol[(size_t)(2*v) * Nc];
      b[2*v+1] = (_Float16)bcol[(size_t)(2*v+1) * Nc];
    }
    acc = __builtin_amdgcn_wmma_f32_16x16x32_f16(false, a, false, b, (short)0, acc,
                                                 false, false);
  }
  float bv = BIAS ? bias[col] : 0.0f;
#pragma unroll
  for (int r = 0; r < 8; ++r) {
    float v = acc[r] + bv;
    if (RELU) v = fmaxf(v, 0.0f);
    C[(size_t)(rt * 16 + r + (half << 3)) * Nc + col] = v;
  }
}

// ---- fold attention vector into weight: p[k,h] = sum_c W[k, h*C+c] * a[h*C+c]
__global__ void k_proj(const float* __restrict__ W, const float* __restrict__ a,
                       float* p, int K, int H, int C){
  int t = blockIdx.x * blockDim.x + threadIdx.x;
  if (t >= K * H) return;
  int k = t / H, h = t - k * H;
  float s = 0.f;
  for (int c = 0; c < C; ++c) s += W[(size_t)k * H * C + h * C + c] * a[h * C + c];
  p[t] = s;
}

// o1[n,h] = X[n,:] @ p1[:,h] ; o2[n,h] = X[n,:] @ p2[:,h]  (single pass over X)
__global__ void k_rowdot_dual(const float* __restrict__ X,
                              const float* __restrict__ p1, const float* __restrict__ p2,
                              float* o1, float* o2, int N, int K, int H){
  int t = blockIdx.x * blockDim.x + threadIdx.x;
  if (t >= N * H) return;
  int n = t / H, h = t - n * H;
  const float* row = X + (size_t)n * K;
  float s1 = 0.f, s2 = 0.f;
  for (int k = 0; k < K; ++k) {
    float xv = row[k];
    s1 += xv * p1[k * H + h];
    s2 += xv * p2[k * H + h];
  }
  o1[t] = s1;
  o2[t] = s2;
}

// aedg[e,h] = ea2[e,:] @ q[:,h]  (self-loop rows come from mean_attr)
__global__ void k_edgedot(const float* __restrict__ ea, const float* __restrict__ mean,
                          const float* __restrict__ q, float* aedg, int E, int E2, int H){
  int t = blockIdx.x * blockDim.x + threadIdx.x;
  if (t >= E2 * H) return;
  int e = t / H, h = t - e * H;
  const float* row = (e < E) ? (ea + (size_t)e * 16) : (mean + (size_t)(e - E) * 16);
  float s = 0.f;
#pragma unroll
  for (int f = 0; f < 16; ++f) s += row[f] * q[f * H + h];
  aedg[t] = s;
}

__global__ void k_passmax(const int* __restrict__ src, const int* __restrict__ dst,
                          const float* __restrict__ asrc, const float* __restrict__ adst,
                          const float* __restrict__ aedg, float* logits, unsigned* m,
                          int E, int E2, int H){
  int t = blockIdx.x * blockDim.x + threadIdx.x;
  if (t >= E2 * H) return;
  int e = t / H, h = t - e * H;
  int s, d;
  if (e < E) { s = src[e]; d = dst[e]; } else { s = d = e - E; }
  float L = lrelu(asrc[s * H + h] + adst[d * H + h] + aedg[t]);
  logits[t] = L;
  atomicMax(&m[d * H + h], mapf(L));
}

__global__ void k_passsum(const int* __restrict__ dst, float* lx,
                          const unsigned* __restrict__ m, float* denom,
                          int E, int E2, int H){
  int t = blockIdx.x * blockDim.x + threadIdx.x;
  if (t >= E2 * H) return;
  int e = t / H, h = t - e * H;
  int d = (e < E) ? dst[e] : e - E;
  float ex = __expf(lx[t] - unmapf(m[d * H + h]));
  lx[t] = ex;                       // reuse logits buffer for exp values
  atomicAdd(&denom[d * H + h], ex);
}

// alpha[e,h] = ex[e,h] / denom[dst,h]  (in place over ex buffer)
__global__ void k_alpha(const int* __restrict__ dst, float* lx,
                        const float* __restrict__ denom, int E, int E2, int H){
  int t = blockIdx.x * blockDim.x + threadIdx.x;
  if (t >= E2 * H) return;
  int e = t / H, h = t - e * H;
  int d = (e < E) ? dst[e] : e - E;
  lx[t] = lx[t] / denom[d * H + h];
}

// acc[dst, h, c..c+3] += alpha[e,h] * xl[src, h, c..c+3]; one b128 load + 4 atomics
__global__ void k_scatter4(const int* __restrict__ src, const int* __restrict__ dst,
                           const float* __restrict__ alpha, const float* __restrict__ xl,
                           float* acc, int E, int E2, int H, int C4){
  int t = blockIdx.x * blockDim.x + threadIdx.x;
  int HC4 = H * C4;
  if (t >= E2 * HC4) return;
  int e  = t / HC4;
  int r  = t - e * HC4;
  int h  = r / C4;
  int c4 = r - h * C4;
  int s, d;
  if (e < E) { s = src[e]; d = dst[e]; } else { s = d = e - E; }
  float a = alpha[e * H + h];
  const float4 xv = *(const float4*)(xl + ((size_t)s * H + h) * (C4 * 4) + c4 * 4);
  float* ap = acc + ((size_t)d * H + h) * (C4 * 4) + c4 * 4;
  atomicAdd(ap + 0, a * xv.x);
  atomicAdd(ap + 1, a * xv.y);
  atomicAdd(ap + 2, a * xv.z);
  atomicAdd(ap + 3, a * xv.w);
}

__global__ void k_biasrelu(float* acc, const float* __restrict__ b, int total, int Nc){
  int t = blockIdx.x * blockDim.x + threadIdx.x;
  if (t >= total) return;
  float v = acc[t] + b[t % Nc];
  acc[t] = fmaxf(v, 0.0f);
}

// out[n] = h2[n,:] @ W_out + b_out  (one wave per node, shuffle reduction)
__global__ void k_final(const float* __restrict__ h2, const float* __restrict__ Wout,
                        const float* __restrict__ bout, float* out, int N){
  int lane = threadIdx.x & 31;
  int node = blockIdx.x * (blockDim.x >> 5) + (threadIdx.x >> 5);
  if (node >= N) return;
  const float* row = h2 + (size_t)node * 64;
  float s = row[lane] * Wout[lane] + row[lane + 32] * Wout[lane + 32];
  for (int off = 16; off; off >>= 1) s += __shfl_xor(s, off, 32);
  if (lane == 0) out[node] = s + bout[0];
}

static inline int cdiv(long long a, long long b){ return (int)((a + b - 1) / b); }

extern "C" void kernel_launch(void* const* d_in, const int* in_sizes, int n_in,
                              void* d_out, int out_size, void* d_ws, size_t ws_size,
                              hipStream_t stream) {
  (void)n_in; (void)out_size; (void)ws_size;
  const float* x       = (const float*)d_in[0];
  const int*   ei      = (const int*)  d_in[1];
  const float* ea      = (const float*)d_in[2];
  const float* W_emb   = (const float*)d_in[3];
  const float* b_emb   = (const float*)d_in[4];
  const float* W1      = (const float*)d_in[5];
  const float* We1     = (const float*)d_in[6];
  const float* a_src1  = (const float*)d_in[7];
  const float* a_dst1  = (const float*)d_in[8];
  const float* a_edge1 = (const float*)d_in[9];
  const float* b1      = (const float*)d_in[10];
  const float* W2      = (const float*)d_in[11];
  const float* We2     = (const float*)d_in[12];
  const float* a_src2  = (const float*)d_in[13];
  const float* a_dst2  = (const float*)d_in[14];
  const float* a_edge2 = (const float*)d_in[15];
  const float* b2      = (const float*)d_in[16];
  const float* W_out   = (const float*)d_in[17];
  const float* b_out   = (const float*)d_in[18];
  float* out = (float*)d_out;

  const int N  = in_sizes[0] / 32;       // 50000
  const int E  = in_sizes[2] / 16;       // 400000
  const int E2 = E + N;                  // with self-loops
  const int* srcp = ei;
  const int* dstp = ei + E;

  float* ws = (float*)d_ws;
  size_t off = 0;
  float*    mean = ws + off; off += (size_t)N * 16;
  float*    cnt  = ws + off; off += (size_t)N;
  // keep all large buffers 16B-aligned (offsets multiples of 4 floats)
  off = (off + 3) & ~(size_t)3;
  float*    h0   = ws + off; off += (size_t)N * 64;    // aliased as xl2 in layer 2
  float*    xl1  = ws + off; off += (size_t)N * 256;
  float*    asrc = ws + off; off += (size_t)N * 4;
  float*    adst = ws + off; off += (size_t)N * 4;
  float*    aedg = ws + off; off += (size_t)E2 * 4;
  float*    lgts = ws + off; off += (size_t)E2 * 4;    // logits -> exp -> alpha in place
  unsigned* mbuf = (unsigned*)(ws + off); off += (size_t)N * 4;
  float*    den  = ws + off; off += (size_t)N * 4;
  float*    acc1 = ws + off; off += (size_t)N * 256;   // becomes h1 in place
  float*    acc2 = ws + off; off += (size_t)N * 64;    // becomes h2 in place
  float*    ps   = ws + off; off += 256;
  float*    pd   = ws + off; off += 256;
  float*    qv   = ws + off; off += 64;
  float*    xl2  = h0;

  const unsigned MAPNEGMAX = 0x00800000u;   // mapf(-FLT_MAX)
  const int T = 256;

  // self-loop mean edge attributes
  k_fill_f32<<<cdiv((long long)N*16,T), T, 0, stream>>>(mean, 0.f, N*16);
  k_fill_f32<<<cdiv(N,T), T, 0, stream>>>(cnt, 0.f, N);
  k_mean_accum<<<cdiv((long long)E*4,T), T, 0, stream>>>(ea, dstp, mean, cnt, E);
  k_mean_div<<<cdiv((long long)N*16,T), T, 0, stream>>>(mean, cnt, N);

  // embedding: h0 = relu(x @ W_emb + b_emb)   [N,32]@[32,64]
  k_wmma_gemm<true,true><<<dim3(N/16, 4), 32, 0, stream>>>(x, W_emb, b_emb, h0, 32, 64);

  // ---- GAT layer 1 (H=4, C=64) ----
  k_proj<<<1, 256, 0, stream>>>(W1, a_src1, ps, 64, 4, 64);
  k_proj<<<1, 256, 0, stream>>>(W1, a_dst1, pd, 64, 4, 64);
  k_proj<<<1, 64,  0, stream>>>(We1, a_edge1, qv, 16, 4, 64);
  k_rowdot_dual<<<cdiv((long long)N*4,T), T, 0, stream>>>(h0, ps, pd, asrc, adst, N, 64, 4);
  k_edgedot<<<cdiv((long long)E2*4,T), T, 0, stream>>>(ea, mean, qv, aedg, E, E2, 4);
  k_wmma_gemm<false,false><<<dim3(N/16, 16), 32, 0, stream>>>(h0, W1, nullptr, xl1, 64, 256);

  k_fill_u32<<<cdiv((long long)N*4,T), T, 0, stream>>>(mbuf, MAPNEGMAX, N*4);
  k_fill_f32<<<cdiv((long long)N*4,T), T, 0, stream>>>(den, 0.f, N*4);
  k_fill_f32<<<cdiv((long long)N*256,T), T, 0, stream>>>(acc1, 0.f, N*256);
  k_passmax<<<cdiv((long long)E2*4,T), T, 0, stream>>>(srcp, dstp, asrc, adst, aedg, lgts, mbuf, E, E2, 4);
  k_passsum<<<cdiv((long long)E2*4,T), T, 0, stream>>>(dstp, lgts, mbuf, den, E, E2, 4);
  k_alpha<<<cdiv((long long)E2*4,T), T, 0, stream>>>(dstp, lgts, den, E, E2, 4);
  k_scatter4<<<cdiv((long long)E2*64,T), T, 0, stream>>>(srcp, dstp, lgts, xl1, acc1, E, E2, 4, 16);
  k_biasrelu<<<cdiv((long long)N*256,T), T, 0, stream>>>(acc1, b1, N*256, 256);

  // ---- GAT layer 2 (H=1, C=64) ----
  k_proj<<<1, 256, 0, stream>>>(W2, a_src2, ps, 256, 1, 64);
  k_proj<<<1, 256, 0, stream>>>(W2, a_dst2, pd, 256, 1, 64);
  k_proj<<<1, 64,  0, stream>>>(We2, a_edge2, qv, 16, 1, 64);
  k_rowdot_dual<<<cdiv(N,T), T, 0, stream>>>(acc1, ps, pd, asrc, adst, N, 256, 1);
  k_edgedot<<<cdiv(E2,T), T, 0, stream>>>(ea, mean, qv, aedg, E, E2, 1);
  k_wmma_gemm<false,false><<<dim3(N/16, 4), 32, 0, stream>>>(acc1, W2, nullptr, xl2, 256, 64);

  k_fill_u32<<<cdiv(N,T), T, 0, stream>>>(mbuf, MAPNEGMAX, N);
  k_fill_f32<<<cdiv(N,T), T, 0, stream>>>(den, 0.f, N);
  k_fill_f32<<<cdiv((long long)N*64,T), T, 0, stream>>>(acc2, 0.f, N*64);
  k_passmax<<<cdiv(E2,T), T, 0, stream>>>(srcp, dstp, asrc, adst, aedg, lgts, mbuf, E, E2, 1);
  k_passsum<<<cdiv(E2,T), T, 0, stream>>>(dstp, lgts, mbuf, den, E, E2, 1);
  k_alpha<<<cdiv(E2,T), T, 0, stream>>>(dstp, lgts, den, E, E2, 1);
  k_scatter4<<<cdiv((long long)E2*16,T), T, 0, stream>>>(srcp, dstp, lgts, xl2, acc2, E, E2, 1, 16);
  k_biasrelu<<<cdiv((long long)N*64,T), T, 0, stream>>>(acc2, b2, N*64, 64);

  // out = h2 @ W_out + b_out
  k_final<<<cdiv(N,8), 256, 0, stream>>>(acc2, W_out, b_out, out, N);
}
